// WindowAttention_30872224924140
// MI455X (gfx1250) — matrix operand order repaired and checked by reading, not compiled, not run on hardware
//
#include <hip/hip_runtime.h>

#define B_WIN 1024
#define NWIN  216
#define NPAD  224
#define HEADS 6
#define DH    32
#define CDIM  192
#define F3    576
#define XS    200   // LDS row stride (u16) for staged activations: conflict-free
#define VSTR  232   // LDS row stride (u16) for vT / P strips: conflict-free
#define QK_SCALE 0.17677669529663687f

typedef __attribute__((ext_vector_type(16))) __bf16        v16bf;
typedef __attribute__((ext_vector_type(8)))  float         v8f;
typedef __attribute__((ext_vector_type(8)))  unsigned short u16x8;
typedef __attribute__((ext_vector_type(4)))  unsigned short u16x4;
typedef __attribute__((ext_vector_type(4)))  float         f32x4;

__device__ __forceinline__ unsigned short bf16_rne(float f){
  unsigned u = __builtin_bit_cast(unsigned, f);
  u += 0x7FFFu + ((u >> 16) & 1u);
  return (unsigned short)(u >> 16);
}

__device__ __forceinline__ v16bf frag_from_halves(u16x8 lo, u16x8 hi){
  v16bf f;
#pragma unroll
  for (int i = 0; i < 8; ++i){
    f[i]     = __builtin_bit_cast(__bf16, (unsigned short)lo[i]);
    f[i + 8] = __builtin_bit_cast(__bf16, (unsigned short)hi[i]);
  }
  return f;
}

// A fragment (16x32 bf16), a(m,k) = base[m*stride + k].
// Lane L: m=L&15, h=L>>4; elems 0..7 <- K=8h..8h+7, elems 8..15 <- K=16+8h..16+8h+7.
__device__ __forceinline__ v16bf load_frag_a(const unsigned short* base, int stride, int lane){
  int m = lane & 15, h = lane >> 4;
  const unsigned short* p = base + m * stride + 8 * h;
  return frag_from_halves(*(const u16x8*)p, *(const u16x8*)(p + 16));
}

// B fragment (32x16 bf16), b(k,n) = base[n*stride + k].
// Lane L: n=L&15, h=L>>4; elems 0..7 <- K=8h..8h+7, elems 8..15 <- K=16+8h..16+8h+7.
__device__ __forceinline__ v16bf load_frag_b(const unsigned short* base, int stride, int lane){
  int n = lane & 15, h = lane >> 4;
  const unsigned short* p = base + n * stride + 8 * h;
  return frag_from_halves(*(const u16x8*)p, *(const u16x8*)(p + 16));
}

__device__ __forceinline__ v8f wmma_bf16(v16bf a, v16bf b, v8f c){
  return __builtin_amdgcn_wmma_f32_16x16x32_bf16(false, a, false, b, (short)0, c, false, false);
}

// ---------------------------------------------------------------- weights->bf16
__global__ __launch_bounds__(256) void convert_weights_kernel(
    const float* __restrict__ qkv_w, const float* __restrict__ proj_w,
    unsigned short* __restrict__ qwbf, unsigned short* __restrict__ pwbf){
  int i = blockIdx.x * 256 + threadIdx.x;
  if (i < F3 * CDIM)   qwbf[i] = bf16_rne(qkv_w[i]);
  if (i < CDIM * CDIM) pwbf[i] = bf16_rne(proj_w[i]);
}

// ---------------------------------------------------------------- QKV GEMM
// One block per window b. Y[224(pad), 576] = Xbf[224,192] @ Wbf[576,192]^T.
// Stores q (pre-scaled), k, v as bf16 [B][H][224][32], zero-padded rows.
__global__ __launch_bounds__(256) void qkv_gemm_kernel(
    const float* __restrict__ x, const unsigned short* __restrict__ wbf,
    unsigned short* __restrict__ qws, unsigned short* __restrict__ kws,
    unsigned short* __restrict__ vws){
  extern __shared__ unsigned short lds[];        // [NPAD][XS] bf16
  const int b = blockIdx.x, tid = threadIdx.x;
  const float* xb = x + (size_t)b * NWIN * CDIM;
  for (int i = tid; i < NPAD * (CDIM / 4); i += 256){
    int r = i / (CDIM / 4);
    int c4 = (i % (CDIM / 4)) * 4;
    f32x4 v = {};
    if (r < NWIN) v = *(const f32x4*)(xb + (size_t)r * CDIM + c4);
    u16x4 o; o[0]=bf16_rne(v.x); o[1]=bf16_rne(v.y); o[2]=bf16_rne(v.z); o[3]=bf16_rne(v.w);
    *(u16x4*)&lds[r * XS + c4] = o;
  }
  __syncthreads();
  const int wave = tid >> 5, lane = tid & 31;
  const int nl = lane & 15, hf = lane >> 4;
  for (int t = wave; t < 14 * 36; t += 8){       // 63 tiles per wave, uniform
    int mt = t % 14, ft = t / 14;
    v8f acc = {};
#pragma unroll
    for (int kk = 0; kk < 6; ++kk){
      v16bf a  = load_frag_a(&lds[(mt * 16) * XS + kk * 32], XS, lane);
      v16bf bb = load_frag_b(wbf + (size_t)(ft * 16) * CDIM + kk * 32, CDIM, lane);
      acc = wmma_bf16(a, bb, acc);
    }
    int fcol  = ft * 16 + nl;
    int which = fcol / CDIM;                      // 0=q 1=k 2=v
    int head  = (fcol % CDIM) / DH;
    int d     = fcol % DH;
    unsigned short* dst = (which == 0) ? qws : (which == 1) ? kws : vws;
    float sc = (which == 0) ? QK_SCALE : 1.0f;
    size_t base = ((size_t)(b * HEADS + head)) * NPAD * DH + d;
#pragma unroll
    for (int r = 0; r < 8; ++r){
      int m = mt * 16 + 8 * hf + r;               // padded rows -> zeros (lds zeroed)
      dst[base + (size_t)m * DH] = bf16_rne(acc[r] * sc);
    }
  }
}

// ---------------------------------------------------------------- fused attention
// One 14-wave block per (b,h). Wave wv owns rows 16wv..16wv+15.
__global__ __launch_bounds__(448) void attn_kernel(
    const unsigned short* __restrict__ qws, const unsigned short* __restrict__ kws,
    const unsigned short* __restrict__ vws, const float* __restrict__ bias_table,
    unsigned short* __restrict__ att){
  extern __shared__ unsigned short lds[];
  unsigned short* vT = lds;                       // [DH][VSTR]
  unsigned short* P  = lds + DH * VSTR;           // [14][16][VSTR] wave-private strips
  const int b = blockIdx.x / HEADS, h = blockIdx.x % HEADS;
  const int tid = threadIdx.x;
  const size_t base = ((size_t)(b * HEADS + h)) * NPAD * DH;
  for (int i = tid; i < NPAD * DH; i += 448){     // transpose V into LDS
    int r = i >> 5, c = i & 31;
    vT[c * VSTR + r] = vws[base + (size_t)r * DH + c];
  }
  __syncthreads();
  const int wv = tid >> 5, lane = tid & 31;
  const int nl = lane & 15, hf = lane >> 4;

  v16bf aq = load_frag_a(qws + base + (size_t)(wv * 16) * DH, DH, lane);
  v8f s[14];
#pragma unroll
  for (int ct = 0; ct < 14; ++ct){                // S = (q*scale) @ k^T
    v16bf bk = load_frag_b(kws + base + (size_t)(ct * 16) * DH, DH, lane);
    v8f z = {};
    s[ct] = wmma_bf16(aq, bk, z);
  }
  int mz[8], my8[8], mx8[8]; bool mok[8];
#pragma unroll
  for (int r = 0; r < 8; ++r){
    int m = wv * 16 + 8 * hf + r;
    mok[r] = (m < NWIN);
    int mm = mok[r] ? m : 0;
    mz[r] = mm / 36; my8[r] = (mm % 36) / 6; mx8[r] = mm % 6;
  }
  float rmax[8];
#pragma unroll
  for (int r = 0; r < 8; ++r) rmax[r] = -1e30f;
#pragma unroll
  for (int ct = 0; ct < 14; ++ct){                // + relative-position bias, mask pads
    int n = ct * 16 + nl;
    bool nok = (n < NWIN);
    int nn = nok ? n : 0;
    int nz = nn / 36, ny = (nn % 36) / 6, nx = nn % 6;
#pragma unroll
    for (int r = 0; r < 8; ++r){
      float val = -1e30f;
      if (nok && mok[r]){
        int idx = (mz[r] - nz + 5) * 17 + (my8[r] - ny + 5) * 11 + (mx8[r] - nx + 5);
        val = s[ct][r] + bias_table[idx * HEADS + h];
      }
      s[ct][r] = val;
      rmax[r] = fmaxf(rmax[r], val);
    }
  }
#pragma unroll
  for (int r = 0; r < 8; ++r){                    // row max across 16-lane half
    float v = rmax[r];
    v = fmaxf(v, __shfl_xor(v, 1)); v = fmaxf(v, __shfl_xor(v, 2));
    v = fmaxf(v, __shfl_xor(v, 4)); v = fmaxf(v, __shfl_xor(v, 8));
    rmax[r] = v;
  }
  float rsum[8];
#pragma unroll
  for (int r = 0; r < 8; ++r) rsum[r] = 0.f;
#pragma unroll
  for (int ct = 0; ct < 14; ++ct)
#pragma unroll
    for (int r = 0; r < 8; ++r){
      float p = __expf(s[ct][r] - rmax[r]);
      s[ct][r] = p; rsum[r] += p;
    }
#pragma unroll
  for (int r = 0; r < 8; ++r){
    float v = rsum[r];
    v += __shfl_xor(v, 1); v += __shfl_xor(v, 2);
    v += __shfl_xor(v, 4); v += __shfl_xor(v, 8);
    rsum[r] = 1.0f / v;
  }
  unsigned short* Pst = P + wv * 16 * VSTR;       // wave-private: no barrier needed
#pragma unroll
  for (int ct = 0; ct < 14; ++ct)
#pragma unroll
    for (int r = 0; r < 8; ++r)
      Pst[(8 * hf + r) * VSTR + ct * 16 + nl] = bf16_rne(s[ct][r] * rsum[r]);
#pragma unroll
  for (int dt = 0; dt < 2; ++dt){                 // out = P @ V
    v8f o = {};
#pragma unroll
    for (int kt = 0; kt < 7; ++kt){
      v16bf a  = load_frag_a(Pst + kt * 32, VSTR, lane);
      v16bf bv = load_frag_b(vT + (dt * 16) * VSTR + kt * 32, VSTR, lane);
      o = wmma_bf16(a, bv, o);
    }
#pragma unroll
    for (int r = 0; r < 8; ++r){
      int m = wv * 16 + 8 * hf + r;
      if (m < NWIN)
        att[((size_t)b * NWIN + m) * CDIM + h * DH + dt * 16 + nl] = bf16_rne(o[r]);
    }
  }
}

// ---------------------------------------------------------------- output projection
__global__ __launch_bounds__(256) void proj_gemm_kernel(
    const unsigned short* __restrict__ att, const unsigned short* __restrict__ pwbf,
    const float* __restrict__ pb, float* __restrict__ out){
  extern __shared__ unsigned short lds[];         // [NPAD][XS]
  const int b = blockIdx.x, tid = threadIdx.x;
  const unsigned short* ab = att + (size_t)b * NWIN * CDIM;
  for (int i = tid; i < NPAD * (CDIM / 8); i += 256){
    int r = i / (CDIM / 8);
    int c8 = (i % (CDIM / 8)) * 8;
    u16x8 v = {};
    if (r < NWIN) v = *(const u16x8*)(ab + (size_t)r * CDIM + c8);
    *(u16x8*)&lds[r * XS + c8] = v;
  }
  __syncthreads();
  const int wave = tid >> 5, lane = tid & 31;
  const int nl = lane & 15, hf = lane >> 4;
  for (int t = wave; t < 14 * 12; t += 8){        // 21 tiles per wave, uniform
    int mt = t % 14, ft = t / 14;
    v8f acc = {};
#pragma unroll
    for (int kk = 0; kk < 6; ++kk){
      v16bf a  = load_frag_a(&lds[(mt * 16) * XS + kk * 32], XS, lane);
      v16bf bb = load_frag_b(pwbf + (size_t)(ft * 16) * CDIM + kk * 32, CDIM, lane);
      acc = wmma_bf16(a, bb, acc);
    }
    int fcol = ft * 16 + nl;
    float bias = pb[fcol];
#pragma unroll
    for (int r = 0; r < 8; ++r){
      int m = mt * 16 + 8 * hf + r;
      if (m < NWIN)
        out[((size_t)b * NWIN + m) * CDIM + fcol] = acc[r] + bias;
    }
  }
}

extern "C" void kernel_launch(void* const* d_in, const int* in_sizes, int n_in,
                              void* d_out, int out_size, void* d_ws, size_t ws_size,
                              hipStream_t stream){
  (void)in_sizes; (void)n_in; (void)out_size; (void)ws_size;
  const float* x      = (const float*)d_in[0];
  const float* qkv_w  = (const float*)d_in[1];
  const float* proj_w = (const float*)d_in[2];
  const float* proj_b = (const float*)d_in[3];
  const float* bias_t = (const float*)d_in[4];
  float* out = (float*)d_out;

  unsigned char* ws = (unsigned char*)d_ws;
  size_t off = 0;
  unsigned short* qwbf = (unsigned short*)(ws + off); off += (size_t)F3 * CDIM * 2;
  unsigned short* pwbf = (unsigned short*)(ws + off); off += (size_t)CDIM * CDIM * 2;
  off = (off + 255) & ~(size_t)255;
  const size_t per = (size_t)B_WIN * HEADS * NPAD * DH * 2;
  unsigned short* qws = (unsigned short*)(ws + off); off += per;
  unsigned short* kws = (unsigned short*)(ws + off); off += per;
  unsigned short* vws = (unsigned short*)(ws + off); off += per;
  unsigned short* att = (unsigned short*)(ws + off); off += (size_t)B_WIN * NWIN * CDIM * 2;

  const int GEMM_SHM = NPAD * XS * 2;                       // 89,600 B
  const int ATTN_SHM = (DH * VSTR + 14 * 16 * VSTR) * 2;    // 118,784 B
  hipFuncSetAttribute((const void*)qkv_gemm_kernel,
                      hipFuncAttributeMaxDynamicSharedMemorySize, GEMM_SHM);
  hipFuncSetAttribute((const void*)attn_kernel,
                      hipFuncAttributeMaxDynamicSharedMemorySize, ATTN_SHM);
  hipFuncSetAttribute((const void*)proj_gemm_kernel,
                      hipFuncAttributeMaxDynamicSharedMemorySize, GEMM_SHM);

  convert_weights_kernel<<<(F3 * CDIM + 255) / 256, 256, 0, stream>>>(qkv_w, proj_w, qwbf, pwbf);
  qkv_gemm_kernel<<<B_WIN, 256, GEMM_SHM, stream>>>(x, qwbf, qws, kws, vws);
  attn_kernel<<<B_WIN * HEADS, 448, ATTN_SHM, stream>>>(qws, kws, vws, bias_t, att);
  proj_gemm_kernel<<<B_WIN, 256, GEMM_SHM, stream>>>(att, pwbf, proj_b, out);
}
